// TwoTowerTripletRNN_35081292874050
// MI455X (gfx1250) — matrix operand order repaired and checked by reading, not compiled, not run on hardware
//
#include <hip/hip_runtime.h>

typedef __attribute__((ext_vector_type(8)))  float        v8f;
typedef __attribute__((ext_vector_type(16))) __bf16       v16bf;
typedef __attribute__((ext_vector_type(4)))  unsigned int v4u;
typedef __attribute__((ext_vector_type(8)))  int          v8i;

#define B_     256
#define E_     256
#define H_     512
#define SQ_    64
#define SP_    256
#define NT_    32      /* H/16 column tiles            */
#define KTX_   8       /* E/32 K-tiles (input proj)    */
#define KTH_   16      /* H/32 K-tiles (recurrent)     */
#define FRAG_  1024    /* bytes per 32x16 B-fragment (32 lanes * 32B) */
#define STEPB_ (NT_ * FRAG_)   /* 32KB of x-proj fragments per (bt,t) step */

/* ---- workspace byte offsets ---- */
#define WQX_OFF  ((size_t)0)
#define WQH_OFF  (WQX_OFF + (size_t)KTX_*NT_*FRAG_)             /* +256KB */
#define WPX_OFF  (WQH_OFF + (size_t)KTH_*NT_*FRAG_)             /* +512KB */
#define WPH_OFF  (WPX_OFF + (size_t)KTX_*NT_*FRAG_)             /* +256KB */
#define XPQ_OFF  (WPH_OFF + (size_t)KTH_*NT_*FRAG_)             /* +512KB */
#define XPP_OFF  (XPQ_OFF + (size_t)(B_/16)*SQ_*NT_*FRAG_)      /* +32MB  */
#define XPN_OFF  (XPP_OFF + (size_t)(B_/16)*SP_*NT_*FRAG_)      /* +128MB */

#define XS_STRIDE 264   /* LDS row stride (bf16 elems) for x-tile  */
#define HS_STRIDE 520   /* LDS row stride (bf16 elems) for h-state */

static __device__ __forceinline__ unsigned short f2bf(float f) {
  unsigned u = __float_as_uint(f);
  u += 0x7FFFu + ((u >> 16) & 1u);          /* round-to-nearest-even */
  return (unsigned short)(u >> 16);
}

union AB { uint4 q[2]; v16bf v; };
union CD { float4 q[2]; v8f   v; };

static __device__ __forceinline__ void wait_tensorcnt0() {
#if __has_builtin(__builtin_amdgcn_s_wait_tensorcnt)
  __builtin_amdgcn_s_wait_tensorcnt(0);
#else
  asm volatile("s_wait_tensorcnt 0x0" ::: "memory");
#endif
}

/* TDM: DMA one contiguous 32KB x-projection block (global -> LDS).
 * D# group0/group1 built in SGPRs; 2-D tile: 4096 x 1 elements of 8 bytes.
 * Inline asm form is portable across both toolchains (bypasses the
 * 5-arg vs 6-arg clang builtin difference). Tracked by TENSORcnt.      */
static __device__ __forceinline__ void tdm_load_step(const unsigned char* gsrc,
                                                     unsigned lds_off) {
  unsigned long long ga = (unsigned long long)(uintptr_t)gsrc;
  v4u g0;
  g0.x = 1u;                                    /* count=1, user descriptor   */
  g0.y = lds_off;                               /* lds_addr (bytes)           */
  g0.z = (unsigned)(ga & 0xFFFFFFFFu);          /* global_addr[31:0]          */
  g0.w = (unsigned)((ga >> 32) & 0x01FFFFFFu)   /* global_addr[56:32]         */
         | 0x80000000u;                         /* type = 2 ("image")         */
  v8i g1;
  g1[0] = (int)(3u << 16);        /* workgroup_mask=0, data_size=3 (8 bytes)  */
  g1[1] = (int)(4096u << 16);     /* tensor_dim0 = 4096 (low 16 -> [31:16])   */
  g1[2] = (int)(1u << 16);        /* tensor_dim0 hi=0 ; tensor_dim1 = 1       */
  g1[3] = (int)(4096u << 16);     /* tensor_dim1 hi=0 ; tile_dim0 = 4096      */
  g1[4] = 1;                      /* tile_dim1 = 1, tile_dim2 = 0             */
  g1[5] = 4096;                   /* tensor_dim0_stride = 4096                */
  g1[6] = 0;
  g1[7] = 0;
  asm volatile("tensor_load_to_lds %0, %1" :: "s"(g0), "s"(g1) : "memory");
}

/* ------------------------------------------------------------------ */
/* Kernel A: pack Wq/Wp (f32 [768][512] row-major) into bf16 WMMA     */
/* B-fragment layout: lane L holds 16 consecutive K values for column */
/* ntile*16+(L&15); lanes 16-31 take the K+16..31 half. 32B per lane. */
/* ------------------------------------------------------------------ */
__global__ void pack_weights_kernel(const float* __restrict__ Wq,
                                    const float* __restrict__ Wp,
                                    unsigned char* __restrict__ ws) {
  int tid  = blockIdx.x * blockDim.x + threadIdx.x;   /* 0..49151 */
  int w    = tid / 24576;                             /* 0=Wq, 1=Wp */
  int r    = tid % 24576;
  int frag = r >> 5;                                  /* 0..767 */
  int lane = r & 31;

  const float* W = w ? Wp : Wq;
  size_t outBase;
  int ktile, ntile, krow, fidx;
  if (frag < KTX_ * NT_) {            /* X part: W rows 0..255   */
    fidx  = frag;
    ktile = frag >> 5; ntile = frag & 31;
    krow  = 0;
    outBase = w ? WPX_OFF : WQX_OFF;
  } else {                            /* H part: W rows 256..767 */
    fidx  = frag - KTX_ * NT_;
    ktile = fidx >> 5; ntile = fidx & 31;
    krow  = E_;
    outBase = w ? WPH_OFF : WQH_OFF;
  }
  unsigned short* dst =
      (unsigned short*)(ws + outBase + (size_t)fidx * FRAG_ + (size_t)lane * 32);
  int kbase = krow + ktile * 32 + ((lane & 16) ? 16 : 0);
  int col   = ntile * 16 + (lane & 15);
#pragma unroll
  for (int m = 0; m < 16; ++m)
    dst[m] = f2bf(W[(size_t)(kbase + m) * H_ + col]);
}

/* ------------------------------------------------------------------ */
/* Kernel B: Xproj[bt,t,nt] = X[:,t,:] @ Wx + b, stored in C-fragment */
/* layout (v8f per lane) so the recurrence loads it as the WMMA       */
/* accumulator directly. Grid (16, S, 4), 256 thr (8 waves).          */
/* ------------------------------------------------------------------ */
__global__ void xproj_kernel(const float* __restrict__ X, int S,
                             const unsigned char* __restrict__ wx,
                             const float* __restrict__ bias,
                             unsigned char* __restrict__ xp) {
  __shared__ unsigned short xs[16 * XS_STRIDE];

  int bt = blockIdx.x, t = blockIdx.y, cg = blockIdx.z;
  int tid = threadIdx.x;

  /* stage 16 tokens x 256 feats into LDS as bf16 */
  {
    int i  = tid >> 4;
    int e0 = (tid & 15) * 16;
    const float* src = X + (((size_t)(bt * 16 + i) * S + t) * E_ + e0);
    const float4* s4 = (const float4*)src;
#pragma unroll
    for (int j = 0; j < 4; ++j) {
      float4 v = s4[j];
      unsigned short* d = &xs[i * XS_STRIDE + e0 + j * 4];
      d[0] = f2bf(v.x); d[1] = f2bf(v.y); d[2] = f2bf(v.z); d[3] = f2bf(v.w);
    }
  }
  __syncthreads();

  int lane = tid & 31;
  int wave = tid >> 5;
  int nt   = cg * 8 + wave;             /* this wave's column tile */
  int row  = lane & 15;
  int kA   = (lane & 16) ? 16 : 0;

  CD acc; acc.v = (v8f)0.0f;
#pragma unroll
  for (int kt = 0; kt < KTX_; ++kt) {
    AB a, b;
    const unsigned short* ap = &xs[row * XS_STRIDE + kt * 32 + kA];
    a.q[0] = *(const uint4*)(ap);
    a.q[1] = *(const uint4*)(ap + 8);
    const uint4* bp =
        (const uint4*)(wx + (size_t)(kt * NT_ + nt) * FRAG_ + (size_t)lane * 32);
    b.q[0] = bp[0];
    b.q[1] = bp[1];
    acc.v = __builtin_amdgcn_wmma_f32_16x16x32_bf16(
        false, a.v, false, b.v, (short)0, acc.v, false, false);
  }
  float bv = bias[nt * 16 + (lane & 15)];
#pragma unroll
  for (int r = 0; r < 8; ++r) acc.v[r] += bv;

  float4* dst = (float4*)(xp + ((size_t)(bt * S + t) * NT_ + nt) * FRAG_ +
                          (size_t)lane * 32);
  dst[0] = acc.q[0];
  dst[1] = acc.q[1];
}

/* ------------------------------------------------------------------ */
/* Kernel C: the recurrence. Grid (16 batch-tiles, 3 towers), 256 thr.*/
/* Each WG owns 16 batch rows; h (bf16) lives in LDS; W_h streams from*/
/* L2 each step. The next step's 32KB x-projection block is DMA'd by  */
/* the Tensor Data Mover into double-buffered LDS while the current   */
/* step's WMMAs run (TENSORcnt-tracked async prefetch).               */
/* ------------------------------------------------------------------ */
__global__ void rnn_towers_kernel(const unsigned char* __restrict__ ws,
                                  float* __restrict__ out) {
  __shared__ unsigned short hs[16 * HS_STRIDE];
  __shared__ __align__(16) unsigned char xbuf[2][STEPB_];

  int bt    = blockIdx.x;
  int tower = blockIdx.y;
  int S     = (tower == 0) ? SQ_ : SP_;
  const unsigned char* wh = ws + ((tower == 0) ? WQH_OFF : WPH_OFF);
  const unsigned char* xpWG =
      ws + ((tower == 0) ? XPQ_OFF : (tower == 1) ? XPP_OFF : XPN_OFF) +
      (size_t)bt * S * STEPB_;

  for (int i = threadIdx.x; i < 16 * HS_STRIDE; i += blockDim.x) hs[i] = 0;

  int lane = threadIdx.x & 31;
  int wave = threadIdx.x >> 5;
  int row  = lane & 15;
  int kA   = (lane & 16) ? 16 : 0;
  int rOff = (lane & 16) ? 8 : 0;

  /* wave 0 kicks off the TDM DMA for step 0 */
  if (threadIdx.x < 32)
    tdm_load_step(xpWG, (unsigned)(uintptr_t)(void*)&xbuf[0][0]);

  __syncthreads();

  CD acc[4];
  int buf = 0;

  for (int t = 0; t < S; ++t) {
    wait_tensorcnt0();            /* issuing wave waits for its DMA   */
    __syncthreads();              /* publish xbuf[buf] to all waves   */

    /* init accumulators from the prefetched x-projection (C-frags)  */
#pragma unroll
    for (int c = 0; c < 4; ++c) {
      int nt = wave * 4 + c;
      const float4* xf =
          (const float4*)(&xbuf[buf][(size_t)nt * FRAG_ + (size_t)lane * 32]);
      acc[c].q[0] = xf[0];
      acc[c].q[1] = xf[1];
    }

    /* async DMA of next step's block into the other buffer           */
    if (t + 1 < S && threadIdx.x < 32)
      tdm_load_step(xpWG + (size_t)(t + 1) * STEPB_,
                    (unsigned)(uintptr_t)(void*)&xbuf[buf ^ 1][0]);

    /* h @ W_h : K = 512 in 16 bf16 K-tiles                           */
#pragma unroll 4
    for (int kt = 0; kt < KTH_; ++kt) {
      AB a;
      const unsigned short* ap = &hs[row * HS_STRIDE + kt * 32 + kA];
      a.q[0] = *(const uint4*)(ap);
      a.q[1] = *(const uint4*)(ap + 8);
#pragma unroll
      for (int c = 0; c < 4; ++c) {
        int nt = wave * 4 + c;
        AB b;
        const uint4* bp = (const uint4*)(wh + (size_t)(kt * NT_ + nt) * FRAG_ +
                                         (size_t)lane * 32);
        b.q[0] = bp[0];
        b.q[1] = bp[1];
        acc[c].v = __builtin_amdgcn_wmma_f32_16x16x32_bf16(
            false, a.v, false, b.v, (short)0, acc[c].v, false, false);
      }
    }

    __syncthreads();  /* all reads of old h complete before overwrite */

    /* tanh in f32, write new h (bf16) back to LDS                    */
#pragma unroll
    for (int c = 0; c < 4; ++c) {
      int col0 = (wave * 4 + c) * 16 + (lane & 15);
#pragma unroll
      for (int r = 0; r < 8; ++r) {
        float v = tanhf(acc[c].v[r]);
        acc[c].v[r] = v;
        hs[(r + rOff) * HS_STRIDE + col0] = f2bf(v);
      }
    }
    __syncthreads();
    buf ^= 1;
  }

  /* final hidden state -> output (f32), tower-major [3][256][512]    */
  float* o = out + (size_t)tower * B_ * H_ + (size_t)bt * 16 * H_;
#pragma unroll
  for (int c = 0; c < 4; ++c) {
    int col = (wave * 4 + c) * 16 + (lane & 15);
#pragma unroll
    for (int r = 0; r < 8; ++r)
      o[(size_t)(r + rOff) * H_ + col] = acc[c].v[r];
  }
}

/* ------------------------------------------------------------------ */
extern "C" void kernel_launch(void* const* d_in, const int* in_sizes, int n_in,
                              void* d_out, int out_size, void* d_ws, size_t ws_size,
                              hipStream_t stream) {
  (void)in_sizes; (void)n_in; (void)out_size; (void)ws_size;
  const float* q   = (const float*)d_in[0];
  const float* pos = (const float*)d_in[1];
  const float* neg = (const float*)d_in[2];
  const float* Wq  = (const float*)d_in[3];
  const float* bq  = (const float*)d_in[4];
  const float* Wp  = (const float*)d_in[5];
  const float* bp  = (const float*)d_in[6];
  unsigned char* ws = (unsigned char*)d_ws;
  float* out = (float*)d_out;

  pack_weights_kernel<<<192, 256, 0, stream>>>(Wq, Wp, ws);

  xproj_kernel<<<dim3(16, SQ_, 4), 256, 0, stream>>>(q,   SQ_, ws + WQX_OFF, bq, ws + XPQ_OFF);
  xproj_kernel<<<dim3(16, SP_, 4), 256, 0, stream>>>(pos, SP_, ws + WPX_OFF, bp, ws + XPP_OFF);
  xproj_kernel<<<dim3(16, SP_, 4), 256, 0, stream>>>(neg, SP_, ws + WPX_OFF, bp, ws + XPN_OFF);

  rnn_towers_kernel<<<dim3(16, 3), 256, 0, stream>>>(ws, out);
}
